// GPSNetContext_77704548319709
// MI455X (gfx1250) — matrix operand order repaired and checked by reading, not compiled, not run on hardware
//
#include <hip/hip_runtime.h>
#include <hip/hip_bf16.h>

// ---------------- CDNA5 (gfx1250) wave32 WMMA types ----------------
typedef __bf16 v16bf __attribute__((ext_vector_type(16)));
typedef __bf16 bf8v  __attribute__((ext_vector_type(8)));   // 16-byte chunk
typedef float  v8f   __attribute__((ext_vector_type(8)));

union ABfrag { v16bf v; bf8v h[2]; };

static constexpr int       N_ = 8192;
static constexpr long long R_ = 131072;
static constexpr int       D_ = 2048;
static constexpr int       H_ = 512;
static constexpr int       F_ = 32;

__device__ __forceinline__ v8f wmma_bf16(v16bf a, v16bf b, v8f c) {
  // D(f32,16x16) = A(16x32 bf16) * B(32x16 bf16) + C
  return __builtin_amdgcn_wmma_f32_16x16x32_bf16(false, a, false, b, (short)0, c,
                                                 false, false);
}

// ---------------- utility kernels ----------------
__global__ void k_fill_f32(float* p, float v, long long n) {
  long long i = (long long)blockIdx.x * 256 + threadIdx.x;
  if (i < n) p[i] = v;
}

__global__ void k_cvt_bf16(const float* __restrict__ src, __bf16* __restrict__ dst,
                           long long n) {
  long long i = (long long)blockIdx.x * 256 + threadIdx.x;
  if (i < n) dst[i] = (__bf16)src[i];
}

// Pack a (K x Nc) row-major fp32 weight into the wave32 WMMA B layout:
// out[((nt*Ktiles + kt)*32 + lane)*16 + j] = B[kt*32 + (j>>3)*16 + (lane>>4)*8 + (j&7)][nt*16 + (lane&15)]
__global__ void k_pack_b(const float* __restrict__ B, __bf16* __restrict__ out,
                         int K, int Nc) {
  long long idx = (long long)blockIdx.x * 256 + threadIdx.x;
  long long total = (long long)K * Nc;
  if (idx >= total) return;
  int j    = (int)(idx & 15);
  int lane = (int)((idx >> 4) & 31);
  long long tile = idx >> 9;
  int Ktiles = K >> 5;
  int kt = (int)(tile % Ktiles);
  int nt = (int)(tile / Ktiles);
  int n = lane & 15, half = lane >> 4;
  int sk = kt * 32 + (j >> 3) * 16 + half * 8 + (j & 7);
  int sn = nt * 16 + n;
  out[idx] = (__bf16)B[(long long)sk * Nc + sn];
}

// ---------------- fused ps*po*pu kernel ----------------
// Wave computes a (16 edges) x (32 H-cols) tile pair: the three A fragments
// (gathered s-rows, o-rows, fp32 union rows cvt'd in-register) are reused
// across two B column tiles -> 6 WMMA per k-step.
__global__ __launch_bounds__(256) void k_fused3(
    const __bf16* __restrict__ instbf, const float* __restrict__ unionF,
    const __bf16* __restrict__ Bws, const __bf16* __restrict__ Bwo,
    const __bf16* __restrict__ Bwu, const float* __restrict__ bs,
    const float* __restrict__ bo, const float* __restrict__ bu,
    const int* __restrict__ rel, __bf16* __restrict__ prod) {
  int lane = threadIdx.x & 31;
  int wv = threadIdx.x >> 5;
  long long tile = (long long)blockIdx.x * 8 + wv;
  int htp = (int)(tile & 15);           // (H/16)/2 = 16 tile-pairs
  long long rt = tile >> 4;
  if (rt >= R_ / 16) return;

  int m = lane & 15, half = lane >> 4;
  long long r = rt * 16 + m;
  long long sR = rel[2 * r], oR = rel[2 * r + 1];
  const __bf16* as = instbf + sR * D_ + half * 8;
  const __bf16* ao = instbf + oR * D_ + half * 8;
  const float*  au = unionF + r * (long long)D_ + half * 8;
  const int Ktiles = D_ / 32;           // 64
  const long long Ts = (long long)Ktiles * 512;  // elements per B column tile
  int ht0 = htp * 2;
  const __bf16* bws = Bws + (long long)ht0 * Ts + lane * 16;
  const __bf16* bwo = Bwo + (long long)ht0 * Ts + lane * 16;
  const __bf16* bwu = Bwu + (long long)ht0 * Ts + lane * 16;

  v8f accS0 = {}, accS1 = {}, accO0 = {}, accO1 = {}, accU0 = {}, accU1 = {};
  for (int kk = 0; kk < D_; kk += 32) {
    ABfrag a, b0, b1;
    // ps: A = inst[s], two B column tiles
    a.h[0] = *(const bf8v*)(as);
    a.h[1] = *(const bf8v*)(as + 16);
    b0.h[0] = *(const bf8v*)(bws);
    b0.h[1] = *(const bf8v*)(bws + 8);
    b1.h[0] = *(const bf8v*)(bws + Ts);
    b1.h[1] = *(const bf8v*)(bws + Ts + 8);
    accS0 = wmma_bf16(a.v, b0.v, accS0);
    accS1 = wmma_bf16(a.v, b1.v, accS1);
    // po: A = inst[o]
    a.h[0] = *(const bf8v*)(ao);
    a.h[1] = *(const bf8v*)(ao + 16);
    b0.h[0] = *(const bf8v*)(bwo);
    b0.h[1] = *(const bf8v*)(bwo + 8);
    b1.h[0] = *(const bf8v*)(bwo + Ts);
    b1.h[1] = *(const bf8v*)(bwo + Ts + 8);
    accO0 = wmma_bf16(a.v, b0.v, accO0);
    accO1 = wmma_bf16(a.v, b1.v, accO1);
    // pu: A = union (fp32 -> bf16 in-register)
    float4 u0 = *(const float4*)(au);
    float4 u1 = *(const float4*)(au + 4);
    float4 u2 = *(const float4*)(au + 16);
    float4 u3 = *(const float4*)(au + 20);
    ABfrag c;
    c.v[0]  = (__bf16)u0.x; c.v[1]  = (__bf16)u0.y; c.v[2]  = (__bf16)u0.z; c.v[3]  = (__bf16)u0.w;
    c.v[4]  = (__bf16)u1.x; c.v[5]  = (__bf16)u1.y; c.v[6]  = (__bf16)u1.z; c.v[7]  = (__bf16)u1.w;
    c.v[8]  = (__bf16)u2.x; c.v[9]  = (__bf16)u2.y; c.v[10] = (__bf16)u2.z; c.v[11] = (__bf16)u2.w;
    c.v[12] = (__bf16)u3.x; c.v[13] = (__bf16)u3.y; c.v[14] = (__bf16)u3.z; c.v[15] = (__bf16)u3.w;
    b0.h[0] = *(const bf8v*)(bwu);
    b0.h[1] = *(const bf8v*)(bwu + 8);
    b1.h[0] = *(const bf8v*)(bwu + Ts);
    b1.h[1] = *(const bf8v*)(bwu + Ts + 8);
    accU0 = wmma_bf16(c.v, b0.v, accU0);
    accU1 = wmma_bf16(c.v, b1.v, accU1);

    as += 32; ao += 32; au += 32;
    bws += 512; bwo += 512; bwu += 512;
    __builtin_prefetch(bws, 0, 1);
  }
  int hc0 = ht0 * 16 + m;
  float vbs0 = bs[hc0], vbo0 = bo[hc0], vbu0 = bu[hc0];
  float vbs1 = bs[hc0 + 16], vbo1 = bo[hc0 + 16], vbu1 = bu[hc0 + 16];
#pragma unroll
  for (int v = 0; v < 8; ++v) {
    long long row = rt * 16 + half * 8 + v;
    float p0 = fmaxf(accS0[v] + vbs0, 0.f) * fmaxf(accO0[v] + vbo0, 0.f) *
               fmaxf(accU0[v] + vbu0, 0.f);
    float p1 = fmaxf(accS1[v] + vbs1, 0.f) * fmaxf(accO1[v] + vbo1, 0.f) *
               fmaxf(accU1[v] + vbu1, 0.f);
    prod[row * H_ + hc0]      = (__bf16)p0;
    prod[row * H_ + hc0 + 16] = (__bf16)p1;
  }
}

// ---------------- generic bf16 WMMA GEMM (16 x 64 output per wave) ----------
// C[M,Nc] = relu?(A[M,K] @ Bpacked + bias); one A fragment feeds 4 WMMA/k-step.
__global__ __launch_bounds__(256) void k_gemm(
    const __bf16* __restrict__ A, const __bf16* __restrict__ Bp,
    const float* __restrict__ bias, float* __restrict__ outF,
    __bf16* __restrict__ outB, const float* __restrict__ rowsumValid,
    int M, int K, int Nc, int do_relu) {
  int lane = threadIdx.x & 31;
  int wv = threadIdx.x >> 5;
  int Ngrp = Nc >> 6;                    // groups of 4 N-tiles
  long long tile = (long long)blockIdx.x * 8 + wv;
  long long totalTiles = (long long)(M >> 4) * Ngrp;
  if (tile >= totalTiles) return;
  int ng = (int)(tile % Ngrp);
  long long mt = tile / Ngrp;

  int m = lane & 15, half = lane >> 4;
  const int Ktiles = K >> 5;
  const long long Ts = (long long)Ktiles * 512;
  const __bf16* aptr = A + (mt * 16 + m) * (long long)K + half * 8;
  const __bf16* bptr = Bp + (long long)(ng * 4) * Ts + lane * 16;

  v8f acc0 = {}, acc1 = {}, acc2 = {}, acc3 = {};
  for (int kk = 0; kk < K; kk += 32) {
    ABfrag a, b0, b1, b2, b3;
    a.h[0] = *(const bf8v*)(aptr);
    a.h[1] = *(const bf8v*)(aptr + 16);
    b0.h[0] = *(const bf8v*)(bptr);
    b0.h[1] = *(const bf8v*)(bptr + 8);
    b1.h[0] = *(const bf8v*)(bptr + Ts);
    b1.h[1] = *(const bf8v*)(bptr + Ts + 8);
    b2.h[0] = *(const bf8v*)(bptr + 2 * Ts);
    b2.h[1] = *(const bf8v*)(bptr + 2 * Ts + 8);
    b3.h[0] = *(const bf8v*)(bptr + 3 * Ts);
    b3.h[1] = *(const bf8v*)(bptr + 3 * Ts + 8);
    acc0 = wmma_bf16(a.v, b0.v, acc0);
    acc1 = wmma_bf16(a.v, b1.v, acc1);
    acc2 = wmma_bf16(a.v, b2.v, acc2);
    acc3 = wmma_bf16(a.v, b3.v, acc3);
    aptr += 32;
    bptr += 512;
    __builtin_prefetch(bptr, 0, 1);
  }
  float bv[4];
  v8f accs[4] = {acc0, acc1, acc2, acc3};
#pragma unroll
  for (int i = 0; i < 4; ++i)
    bv[i] = bias ? bias[(ng * 4 + i) * 16 + m] : 0.f;
#pragma unroll
  for (int v = 0; v < 8; ++v) {
    long long row = mt * 16 + half * 8 + v;
    bool zeroRow = (rowsumValid && rowsumValid[row] == 0.0f);
#pragma unroll
    for (int i = 0; i < 4; ++i) {
      float x = accs[i][v] + bv[i];
      if (do_relu) x = fmaxf(x, 0.f);
      if (zeroRow) x = 0.f;
      long long oidx = row * Nc + (ng * 4 + i) * 16 + m;
      if (outB) outB[oidx] = (__bf16)x;
      else      outF[oidx] = x;
    }
  }
}

// ---------------- gate = relu(prod @ gw + gb).mean(1), global max ----------------
__global__ __launch_bounds__(256) void k_gate(const __bf16* __restrict__ prod,
                                              const float* __restrict__ gw,
                                              const float* __restrict__ gb,
                                              float* __restrict__ gate,
                                              float* __restrict__ gmax) {
  __shared__ float sgw[128 * F_];  // 16 KB chunk of gw
  long long r = (long long)blockIdx.x * 256 + threadIdx.x;
  float acc[F_];
#pragma unroll
  for (int f = 0; f < F_; ++f) acc[f] = 0.f;
  for (int h0 = 0; h0 < H_; h0 += 128) {
    for (int t = threadIdx.x; t < 128 * F_; t += 256) sgw[t] = gw[h0 * F_ + t];
    __syncthreads();
    const __bf16* pr = prod + r * H_ + h0;
    for (int hh = 0; hh < 128; ++hh) {
      float p = (float)pr[hh];
      const float* wrow = &sgw[hh * F_];
#pragma unroll
      for (int f = 0; f < F_; ++f) acc[f] += p * wrow[f];
    }
    __syncthreads();
  }
  float s = 0.f;
#pragma unroll
  for (int f = 0; f < F_; ++f) s += fmaxf(acc[f] + gb[f], 0.f);
  float g = s * (1.0f / (float)F_);
  gate[r] = g;
  atomicMax((int*)gmax, __float_as_int(g));  // g >= 0: int compare valid
}

// ---------------- per-edge exp + row sums ----------------
__global__ void k_expg(const float* __restrict__ gate, const float* __restrict__ gmax,
                       const int* __restrict__ rel, float* __restrict__ expg,
                       float* __restrict__ rowsum) {
  long long i = (long long)blockIdx.x * 256 + threadIdx.x;
  if (i >= R_) return;
  float e = expf(gate[i] - gmax[0]);
  expg[i] = e;
  atomicAdd(&rowsum[rel[2 * i]], e);
}

// ---------------- SpMM scatter: mfeat = [atten@msg, atten^T@msg] ----------------
__global__ __launch_bounds__(256) void k_scatter(
    const float* __restrict__ expg, const float* __restrict__ rowsum,
    const int* __restrict__ rel, const float* __restrict__ msg,
    float* __restrict__ mfeat) {
  long long e = blockIdx.x;
  int s = rel[2 * e], o = rel[2 * e + 1];
  float a = expg[e] / (rowsum[s] + 1e-6f);
  const float* mo = msg + (long long)o * (D_ / 2);
  const float* ms = msg + (long long)s * (D_ / 2);
  float* fs = mfeat + (long long)s * D_;            // first half of row s
  float* fo = mfeat + (long long)o * D_ + D_ / 2;   // second half of row o
  for (int c = threadIdx.x; c < D_ / 2; c += 256) {
    atomicAdd(&fs[c], a * mo[c]);
    atomicAdd(&fo[c], a * ms[c]);
  }
}

// ---------------- LayerNorm + relu -> bf16 ----------------
__global__ __launch_bounds__(256) void k_ln(const float* __restrict__ h,
                                            const float* __restrict__ g,
                                            const float* __restrict__ b,
                                            __bf16* __restrict__ out) {
  __shared__ float red[256];
  int tid = threadIdx.x;
  long long row = blockIdx.x;
  const float* hr = h + row * (D_ / 4);  // 512 wide
  float x0 = hr[tid], x1 = hr[tid + 256];
  red[tid] = x0 + x1;
  __syncthreads();
  for (int s = 128; s > 0; s >>= 1) {
    if (tid < s) red[tid] += red[tid + s];
    __syncthreads();
  }
  float mu = red[0] * (1.f / 512.f);
  __syncthreads();
  float d0 = x0 - mu, d1 = x1 - mu;
  red[tid] = d0 * d0 + d1 * d1;
  __syncthreads();
  for (int s = 128; s > 0; s >>= 1) {
    if (tid < s) red[tid] += red[tid + s];
    __syncthreads();
  }
  float rs = rsqrtf(red[0] * (1.f / 512.f) + 1e-5f);
  out[row * 512 + tid]       = (__bf16)fmaxf(d0 * rs * g[tid] + b[tid], 0.f);
  out[row * 512 + tid + 256] = (__bf16)fmaxf(d1 * rs * g[tid + 256] + b[tid + 256], 0.f);
}

// ---------------- host-side orchestration ----------------
extern "C" void kernel_launch(void* const* d_in, const int* in_sizes, int n_in,
                              void* d_out, int out_size, void* d_ws, size_t ws_size,
                              hipStream_t stream) {
  const float* inst  = (const float*)d_in[0];   // (N, D)
  const float* uni   = (const float*)d_in[1];   // (R, D)
  const float* ws_w  = (const float*)d_in[2];   // (D, H)
  const float* ws_b  = (const float*)d_in[3];
  const float* wo_w  = (const float*)d_in[4];
  const float* wo_b  = (const float*)d_in[5];
  const float* wu_w  = (const float*)d_in[6];
  const float* wu_b  = (const float*)d_in[7];
  const float* gw    = (const float*)d_in[8];   // (H, F)
  const float* gb    = (const float*)d_in[9];
  const float* msg_w = (const float*)d_in[10];  // (D, D/2)
  const float* msg_b = (const float*)d_in[11];
  const float* o1_w  = (const float*)d_in[12];  // (D, D/4)
  const float* o1_b  = (const float*)d_in[13];
  const float* ln_g  = (const float*)d_in[14];
  const float* ln_b  = (const float*)d_in[15];
  const float* o2_w  = (const float*)d_in[16];  // (D/4, H)
  const float* o2_b  = (const float*)d_in[17];
  const float* fuse_w = (const float*)d_in[18]; // (H, D)
  const float* fuse_b = (const float*)d_in[19];
  const int* rel     = (const int*)d_in[20];    // (R, 2)
  float* out = (float*)d_out;                   // (N, D)

  unsigned char* w = (unsigned char*)d_ws;
  auto alloc = [&](size_t bytes) {
    void* p = (void*)w;
    w += (bytes + 255) & ~(size_t)255;
    return p;
  };
  __bf16* instbf  = (__bf16*)alloc((size_t)N_ * D_ * 2);
  __bf16* Bws     = (__bf16*)alloc((size_t)D_ * H_ * 2);
  __bf16* Bwo     = (__bf16*)alloc((size_t)D_ * H_ * 2);
  __bf16* Bwu     = (__bf16*)alloc((size_t)D_ * H_ * 2);
  __bf16* Bmsg    = (__bf16*)alloc((size_t)D_ * (D_ / 2) * 2);
  __bf16* Bo1     = (__bf16*)alloc((size_t)D_ * (D_ / 4) * 2);
  __bf16* Bo2     = (__bf16*)alloc((size_t)(D_ / 4) * H_ * 2);
  __bf16* Bfuse   = (__bf16*)alloc((size_t)H_ * D_ * 2);
  __bf16* prod    = (__bf16*)alloc((size_t)R_ * H_ * 2);
  float*  gate    = (float*)alloc((size_t)R_ * 4);
  float*  expg    = (float*)alloc((size_t)R_ * 4);
  float*  gmax    = (float*)alloc(256);
  float*  rowsum  = (float*)alloc((size_t)N_ * 4);
  float*  msg     = (float*)alloc((size_t)N_ * (D_ / 2) * 4);
  float*  mfeat   = (float*)alloc((size_t)N_ * D_ * 4);
  __bf16* mfeatbf = (__bf16*)alloc((size_t)N_ * D_ * 2);
  float*  hbuf    = (float*)alloc((size_t)N_ * (D_ / 4) * 4);
  __bf16* hlnbf   = (__bf16*)alloc((size_t)N_ * (D_ / 4) * 2);
  __bf16* h2bf    = (__bf16*)alloc((size_t)N_ * H_ * 2);
  (void)ws_size; (void)in_sizes; (void)n_in; (void)out_size;

  // --- zero-init accumulators (deterministic per call) ---
  k_fill_f32<<<1, 256, 0, stream>>>(gmax, 0.f, 1);
  k_fill_f32<<<(N_ + 255) / 256, 256, 0, stream>>>(rowsum, 0.f, N_);
  k_fill_f32<<<(int)(((long long)N_ * D_ + 255) / 256), 256, 0, stream>>>(
      mfeat, 0.f, (long long)N_ * D_);

  // --- bf16 conversions / weight packing ---
  k_cvt_bf16<<<(int)(((long long)N_ * D_ + 255) / 256), 256, 0, stream>>>(
      inst, instbf, (long long)N_ * D_);
  auto pack = [&](const float* src, __bf16* dst, int K, int Nc) {
    long long total = (long long)K * Nc;
    k_pack_b<<<(int)((total + 255) / 256), 256, 0, stream>>>(src, dst, K, Nc);
  };
  pack(ws_w, Bws, D_, H_);
  pack(wo_w, Bwo, D_, H_);
  pack(wu_w, Bwu, D_, H_);
  pack(msg_w, Bmsg, D_, D_ / 2);
  pack(o1_w, Bo1, D_, D_ / 4);
  pack(o2_w, Bo2, D_ / 4, H_);
  pack(fuse_w, Bfuse, H_, D_);

  // --- prod = relu(ps)*relu(po)*relu(pu) : 3 gathered GEMMs fused ---
  {
    long long tiles = (R_ / 16) * (H_ / 32);  // 131072 wave-tiles (16x32 each)
    k_fused3<<<(int)(tiles / 8), 256, 0, stream>>>(instbf, uni, Bws, Bwo, Bwu,
                                                   ws_b, wo_b, wu_b, rel, prod);
  }

  // --- gate + global max ---
  k_gate<<<(int)(R_ / 256), 256, 0, stream>>>(prod, gw, gb, gate, gmax);

  // --- exp + row sums (sparse softmax) ---
  k_expg<<<(int)(R_ / 256), 256, 0, stream>>>(gate, gmax, rel, expg, rowsum);

  // --- msg = inst @ msg_w + msg_b ---
  {
    long long tiles = (long long)(N_ / 16) * ((D_ / 2) / 64);
    k_gemm<<<(int)(tiles / 8), 256, 0, stream>>>(instbf, Bmsg, msg_b, msg,
                                                 nullptr, nullptr, N_, D_,
                                                 D_ / 2, 0);
  }

  // --- mfeat = [atten@msg, atten^T@msg] via edge scatter ---
  k_scatter<<<(int)R_, 256, 0, stream>>>(expg, rowsum, rel, msg, mfeat);

  // --- h = mfeat @ o1_w + o1_b ---
  k_cvt_bf16<<<(int)(((long long)N_ * D_ + 255) / 256), 256, 0, stream>>>(
      mfeat, mfeatbf, (long long)N_ * D_);
  {
    long long tiles = (long long)(N_ / 16) * ((D_ / 4) / 64);
    k_gemm<<<(int)(tiles / 8), 256, 0, stream>>>(mfeatbf, Bo1, o1_b, hbuf,
                                                 nullptr, nullptr, N_, D_,
                                                 D_ / 4, 0);
  }

  // --- LayerNorm + inner relu -> bf16 ---
  k_ln<<<N_, 256, 0, stream>>>(hbuf, ln_g, ln_b, hlnbf);

  // --- h2 = relu(hln @ o2_w + o2_b), masked by valid rows -> bf16 ---
  {
    long long tiles = (long long)(N_ / 16) * (H_ / 64);
    k_gemm<<<(int)(tiles / 8), 256, 0, stream>>>(hlnbf, Bo2, o2_b, nullptr,
                                                 h2bf, rowsum, N_, D_ / 4, H_,
                                                 1);
  }

  // --- out = relu(h2 @ fuse_w + fuse_b) ---
  {
    long long tiles = (long long)(N_ / 16) * (D_ / 64);
    k_gemm<<<(int)(tiles / 8), 256, 0, stream>>>(h2bf, Bfuse, fuse_b, out,
                                                 nullptr, nullptr, N_, H_, D_,
                                                 1);
  }
}
